// TranslationModel_32298154066677
// MI455X (gfx1250) — matrix operand order repaired and checked by reading, not compiled, not run on hardware
//
#include <hip/hip_runtime.h>
#include <hip/hip_bf16.h>
#include <stdint.h>

typedef __attribute__((ext_vector_type(16))) _Float16 v16h;
typedef __attribute__((ext_vector_type(8)))  _Float16 v8h;
typedef __attribute__((ext_vector_type(8)))  float    v8f;
typedef __attribute__((ext_vector_type(4)))  uint32_t v4u;
typedef __attribute__((ext_vector_type(8)))  int      v8i;
typedef __attribute__((ext_vector_type(4)))  int      v4i;

#define DEV __device__ __forceinline__

// ---- WMMA helpers (CDNA5 16x16x32 f16 -> f32) -------------------------------
DEV v8f wmma_f32(v16h a, v16h b, v8f c) {
  return __builtin_amdgcn_wmma_f32_16x16x32_f16(false, a, false, b, (short)0, c,
                                                false, false);
}

// A-matrix 16x32 f16 fragment, per ISA 7.12.2:
// lane L: row = m0 + L%16 ; VGPR0..3 hold K = k0 + (L<16?0:8)+{0..7},
//         VGPR4..7 hold K = k0+16 + (L<16?0:8)+{0..7}
DEV v16h load_a_frag(const _Float16* base, int ld, int m0, int k0) {
  int lane = threadIdx.x & 31;
  const _Float16* p = base + (size_t)(m0 + (lane & 15)) * ld + k0 + ((lane >> 4) << 3);
  v8h lo = *(const v8h*)p;
  v8h hi = *(const v8h*)(p + 16);
  v16h r;
#pragma unroll
  for (int i = 0; i < 8; i++) { r[i] = lo[i]; r[8 + i] = hi[i]; }
  return r;
}

// B-matrix 32x16 f16 fragment.  W is stored [N][K] row-major so B[k][n] = W[n][k]
// lane L: col n = n0 + L%16 ; K = k0 + (L<16?0:8)+{0..7}, then k0+16 + (L<16?0:8)+{0..7}
DEV v16h load_b_frag(const _Float16* W, int ld, int n0, int k0) {
  int lane = threadIdx.x & 31;
  const _Float16* p = W + (size_t)(n0 + (lane & 15)) * ld + k0 + ((lane >> 4) << 3);
  v8h lo = *(const v8h*)p;
  v8h hi = *(const v8h*)(p + 16);
  v16h r;
#pragma unroll
  for (int i = 0; i < 8; i++) { r[i] = lo[i]; r[8 + i] = hi[i]; }
  return r;
}

DEV float sigm(float x) { return 1.f / (1.f + __expf(-x)); }
DEV float tanh_(float x) { float e = __expf(2.f * x); return (e - 1.f) / (e + 1.f); }

// ---- small utility kernels --------------------------------------------------
__global__ void k_f32_to_f16(const float* __restrict__ in, _Float16* __restrict__ out, int n) {
  int i = blockIdx.x * blockDim.x + threadIdx.x;
  int stride = gridDim.x * blockDim.x;
  for (; i < n; i += stride) out[i] = (_Float16)in[i];
}

__global__ void k_zero_u64(unsigned long long* __restrict__ p, int n) {
  int i = blockIdx.x * blockDim.x + threadIdx.x;
  int stride = gridDim.x * blockDim.x;
  for (; i < n; i += stride) p[i] = 0ull;
}

// gather encoder embeddings, time-major rows (t*64+b), f32 -> f16
__global__ void k_gather_emb(const int* __restrict__ x, const float* __restrict__ emb,
                             _Float16* __restrict__ out) {
  int p = blockIdx.x;              // p = t*64 + b
  int t = p >> 6, b = p & 63;
  int row = x[b * 64 + t];
  const float* src = emb + (size_t)row * 512;
  _Float16* dst = out + (size_t)p * 512;
  for (int i = threadIdx.x; i < 512; i += blockDim.x) dst[i] = (_Float16)src[i];
}

// ---- encoder input projection: Xg[dir] = Emb16 @ Wih^T  ([4096,512]x[512,1024])
__global__ __launch_bounds__(256) void k_xg(const _Float16* __restrict__ emb16,
                                            const _Float16* __restrict__ wihF,
                                            const _Float16* __restrict__ wihB,
                                            float* __restrict__ xg) {
  int dir = blockIdx.z;
  const _Float16* W = dir ? wihB : wihF;
  float* out = xg + (size_t)dir * 4096 * 1024;
  int wid = threadIdx.x >> 5, lane = threadIdx.x & 31;
  int row0 = blockIdx.x * 64;
  int n0 = blockIdx.y * 128 + wid * 16;
  v8f acc[4] = {};
  for (int kc = 0; kc < 512; kc += 32) {
    v16h bfr = load_b_frag(W, 512, n0, kc);
#pragma unroll
    for (int m = 0; m < 4; m++)
      acc[m] = wmma_f32(load_a_frag(emb16, 512, row0 + m * 16, kc), bfr, acc[m]);
  }
  int col = n0 + (lane & 15);
  int rb = row0 + ((lane >> 4) << 3);
#pragma unroll
  for (int m = 0; m < 4; m++)
#pragma unroll
    for (int v = 0; v < 8; v++)
      out[(size_t)(rb + m * 16 + v) * 1024 + col] = acc[m][v];
}

// ---- encoder recurrence: one workgroup per direction, h in LDS, c in VGPRs --
__global__ __launch_bounds__(1024) void k_enc_rnn(const float* __restrict__ xg,
                                                  const _Float16* __restrict__ whhF,
                                                  const _Float16* __restrict__ whhB,
                                                  const float* __restrict__ bF,
                                                  const float* __restrict__ bB,
                                                  _Float16* __restrict__ hbuf0,
                                                  float* __restrict__ cdec) {
  __shared__ _Float16 hsh[64 * 256];   // 32 KB
  int dir = blockIdx.x;
  const _Float16* Whh = dir ? whhB : whhF;
  const float* bias = dir ? bB : bF;
  const float* Xg = xg + (size_t)dir * 4096 * 1024;
  int wid = threadIdx.x >> 5, lane = threadIdx.x & 31;

  for (int i = threadIdx.x; i < 64 * 256; i += blockDim.x) hsh[i] = (_Float16)0.f;
  float creg[2][8];
#pragma unroll
  for (int cb = 0; cb < 2; cb++)
#pragma unroll
    for (int v = 0; v < 8; v++) creg[cb][v] = 0.f;
  __syncthreads();

  for (int step = 0; step < 64; step++) {
    int trow = dir ? (63 - step) : step;
    const float* Xrow = Xg + (size_t)trow * 64 * 1024;
    v8f acc[2][4];
#pragma unroll
    for (int cb = 0; cb < 2; cb++) {
      int combo = wid * 2 + cb;
      int m = combo & 3, u = combo >> 2;           // u in 0..15
#pragma unroll
      for (int g = 0; g < 4; g++) {
        int col = g * 256 + u * 16 + (lane & 15);
#pragma unroll
        for (int v = 0; v < 8; v++)
          acc[cb][g][v] = Xrow[(size_t)(m * 16 + v + ((lane >> 4) << 3)) * 1024 + col] + bias[col];
      }
      for (int kc = 0; kc < 256; kc += 32) {
        v16h afr = load_a_frag(hsh, 256, m * 16, kc);
#pragma unroll
        for (int g = 0; g < 4; g++)
          acc[cb][g] = wmma_f32(afr, load_b_frag(Whh, 256, g * 256 + u * 16, kc), acc[cb][g]);
      }
    }
    __syncthreads();   // all waves finished reading old h
#pragma unroll
    for (int cb = 0; cb < 2; cb++) {
      int combo = wid * 2 + cb;
      int m = combo & 3, u = combo >> 2;
#pragma unroll
      for (int v = 0; v < 8; v++) {
        float cn = sigm(acc[cb][1][v]) * creg[cb][v] + sigm(acc[cb][0][v]) * tanh_(acc[cb][2][v]);
        float hn = sigm(acc[cb][3][v]) * tanh_(cn);
        creg[cb][v] = cn;
        int row = m * 16 + v + ((lane >> 4) << 3);
        hsh[row * 256 + u * 16 + (lane & 15)] = (_Float16)hn;
      }
    }
    __syncthreads();
  }
  // final states -> decoder init (forward dir = cols 0..255, backward = 256..511)
  for (int i = threadIdx.x; i < 64 * 256; i += blockDim.x) {
    int row = i >> 8, col = i & 255;
    hbuf0[row * 512 + dir * 256 + col] = hsh[i];
  }
#pragma unroll
  for (int cb = 0; cb < 2; cb++) {
    int combo = wid * 2 + cb;
    int m = combo & 3, u = combo >> 2;
#pragma unroll
    for (int v = 0; v < 8; v++) {
      int row = m * 16 + v + ((lane >> 4) << 3);
      int col = u * 16 + (lane & 15);
      cdec[row * 512 + dir * 256 + col] = creg[cb][v];
    }
  }
}

// ---- decoder LSTM cell step (16 blocks x 8 waves = 128 output tiles) --------
__global__ __launch_bounds__(256) void k_dec_step(int step,
                                                  const float* __restrict__ dec_emb,
                                                  const _Float16* __restrict__ dwih,
                                                  const _Float16* __restrict__ dwhh,
                                                  const float* __restrict__ dbias,
                                                  const unsigned long long* __restrict__ amax,
                                                  _Float16* __restrict__ hbuf,
                                                  float* __restrict__ cdec) {
  int wid = threadIdx.x >> 5, lane = threadIdx.x & 31;
  int gid = blockIdx.x * 8 + wid;     // 0..127
  int m = gid & 3, u = gid >> 2;      // u 0..31
  const _Float16* hA = hbuf + (size_t)(step & 1) * 64 * 512;       // read
  _Float16* hB = hbuf + (size_t)((step + 1) & 1) * 64 * 512;       // write

  // per-lane token id from previous step's argmax (START_IDX=1 at step 0)
  int brow = m * 16 + (lane & 15);
  int y = 1;
  if (step > 0)
    y = (int)(0xFFFFFFFFu - (uint32_t)(amax[(size_t)(step - 1) * 64 + brow] & 0xFFFFFFFFull));
  const float* er = dec_emb + (size_t)y * 512;
  int kh = (lane >> 4) << 3;

  v8f acc[4] = {};
  // input part: e @ dec_wih^T  (gather+convert f32 embedding rows on the fly)
  for (int kc = 0; kc < 512; kc += 32) {
    v16h afr;
    const float* p0 = er + kc + kh;
    const float* p1 = er + kc + 16 + kh;
#pragma unroll
    for (int i = 0; i < 8; i++) { afr[i] = (_Float16)p0[i]; afr[8 + i] = (_Float16)p1[i]; }
#pragma unroll
    for (int g = 0; g < 4; g++)
      acc[g] = wmma_f32(afr, load_b_frag(dwih, 512, g * 512 + u * 16, kc), acc[g]);
  }
  // recurrent part: h @ dec_whh^T
  for (int kc = 0; kc < 512; kc += 32) {
    v16h afr = load_a_frag(hA, 512, m * 16, kc);
#pragma unroll
    for (int g = 0; g < 4; g++)
      acc[g] = wmma_f32(afr, load_b_frag(dwhh, 512, g * 512 + u * 16, kc), acc[g]);
  }
  int col = u * 16 + (lane & 15);
#pragma unroll
  for (int v = 0; v < 8; v++) {
    int row = m * 16 + v + kh;
    float gi = acc[0][v] + dbias[0 * 512 + col];
    float gf = acc[1][v] + dbias[1 * 512 + col];
    float gg = acc[2][v] + dbias[2 * 512 + col];
    float go = acc[3][v] + dbias[3 * 512 + col];
    float c = cdec[row * 512 + col];
    float cn = sigm(gf) * c + sigm(gi) * tanh_(gg);
    float hn = sigm(go) * tanh_(cn);
    cdec[row * 512 + col] = cn;
    hB[row * 512 + col] = (_Float16)hn;
  }
}

// ---- classifier + argmax: [64,512] x [512,32000], TDM stages h into LDS -----
__global__ __launch_bounds__(256) void k_clf(int step,
                                             const _Float16* __restrict__ hbuf,
                                             const _Float16* __restrict__ clfw,
                                             const float* __restrict__ clfb,
                                             float* __restrict__ out,
                                             unsigned long long* __restrict__ amax) {
  __shared__ _Float16 Ash[64 * 512];  // 64 KB activation tile
  __shared__ float Lsh[64 * 128];     // 32 KB logits staging
  const _Float16* hA = hbuf + (size_t)((step + 1) & 1) * 64 * 512;

  // Tensor Data Mover: 2D tile (512 x 64 elem, 2B) global -> LDS, wave 0 issues
  if (threadIdx.x < 32) {
    uint64_t ga = (uint64_t)(uintptr_t)hA;
    uint32_t lds = (uint32_t)(uintptr_t)Ash;     // LDS aperture: low 32 bits = LDS addr
    v4u g0;
    g0[0] = 1u;                                   // count=1 valid descriptor
    g0[1] = lds;                                  // lds_addr
    g0[2] = (uint32_t)ga;                         // global_addr[31:0]
    g0[3] = (uint32_t)((ga >> 32) & 0x01FFFFFFu) | (2u << 30);  // addr[56:32] | type=2
    v8i g1;
    g1[0] = (int)(1u << 16);        // data_size = 2 bytes
    g1[1] = (int)(512u << 16);      // tensor_dim0 = 512
    g1[2] = (int)(64u << 16);       // tensor_dim1 = 64
    g1[3] = (int)(512u << 16);      // tile_dim0 = 512
    g1[4] = 64;                     // tile_dim1 = 64 (tile_dim2 = 0)
    g1[5] = 512;                    // tensor_dim0_stride = 512
    g1[6] = 0; g1[7] = 0;
    v4i gz4; gz4[0] = 0; gz4[1] = 0; gz4[2] = 0; gz4[3] = 0;
    v8i gz8;
#pragma unroll
    for (int i = 0; i < 8; i++) gz8[i] = 0;
    // 6-arg flavor on this toolchain: (g0, g1, g2, g3, g4, cpol)
    __builtin_amdgcn_tensor_load_to_lds(g0, g1, gz4, gz4, gz8, 0);
    __builtin_amdgcn_s_wait_tensorcnt(0);
  }
  __syncthreads();

  int wid = threadIdx.x >> 5, lane = threadIdx.x & 31;
  int n0 = blockIdx.x * 128 + wid * 16;
  v8f acc[4] = {};
  for (int kc = 0; kc < 512; kc += 32) {
    v16h bfr = load_b_frag(clfw, 512, n0, kc);
#pragma unroll
    for (int m = 0; m < 4; m++)
      acc[m] = wmma_f32(load_a_frag(Ash, 512, m * 16, kc), bfr, acc[m]);
  }
  float cb = clfb[n0 + (lane & 15)];
  int lcol = wid * 16 + (lane & 15);
#pragma unroll
  for (int m = 0; m < 4; m++)
#pragma unroll
    for (int v = 0; v < 8; v++)
      Lsh[(m * 16 + v + ((lane >> 4) << 3)) * 128 + lcol] = acc[m][v] + cb;
  __syncthreads();

  // write logits to d_out[b][v][t] and reduce block-local argmax
  for (int e = threadIdx.x; e < 64 * 128; e += blockDim.x) {
    int b = e >> 7, cl = e & 127;
    int v = blockIdx.x * 128 + cl;
    out[((size_t)b * 32000 + v) * 63 + step] = Lsh[e];
  }
  if (threadIdx.x < 64) {
    int b = threadIdx.x;
    unsigned long long best = 0ull;
    for (int cl = 0; cl < 128; cl++) {
      float f = Lsh[b * 128 + cl];
      uint32_t bits = __float_as_uint(f);
      uint32_t key = (bits & 0x80000000u) ? ~bits : (bits | 0x80000000u);
      unsigned long long pk =
          ((unsigned long long)key << 32) |
          (unsigned long long)(0xFFFFFFFFu - (uint32_t)(blockIdx.x * 128 + cl));
      best = pk > best ? pk : best;
    }
    atomicMax(&amax[(size_t)step * 64 + b], best);
  }
}

// ---- host-side orchestration ------------------------------------------------
extern "C" void kernel_launch(void* const* d_in, const int* in_sizes, int n_in,
                              void* d_out, int out_size, void* d_ws, size_t ws_size,
                              hipStream_t stream) {
  const int*   x         = (const int*)  d_in[0];
  const float* enc_emb   = (const float*)d_in[1];
  const float* enc_wih_f = (const float*)d_in[2];
  const float* enc_whh_f = (const float*)d_in[3];
  const float* enc_b_f   = (const float*)d_in[4];
  const float* enc_wih_b = (const float*)d_in[5];
  const float* enc_whh_b = (const float*)d_in[6];
  const float* enc_b_b   = (const float*)d_in[7];
  const float* dec_emb   = (const float*)d_in[8];
  const float* dec_wih   = (const float*)d_in[9];
  const float* dec_whh   = (const float*)d_in[10];
  const float* dec_b     = (const float*)d_in[11];
  const float* clf_w     = (const float*)d_in[12];
  const float* clf_b     = (const float*)d_in[13];
  float* out = (float*)d_out;

  char* ws = (char*)d_ws;
  size_t off = 0;
  auto alloc = [&](size_t bytes) -> char* {
    char* p = ws + off;
    off = (off + bytes + 255) & ~(size_t)255;
    return p;
  };
  _Float16* clfw16  = (_Float16*)alloc(32000ull * 512 * 2);
  _Float16* dwih16  = (_Float16*)alloc(2048ull * 512 * 2);
  _Float16* dwhh16  = (_Float16*)alloc(2048ull * 512 * 2);
  _Float16* ewihf16 = (_Float16*)alloc(1024ull * 512 * 2);
  _Float16* ewhhf16 = (_Float16*)alloc(1024ull * 256 * 2);
  _Float16* ewihb16 = (_Float16*)alloc(1024ull * 512 * 2);
  _Float16* ewhhb16 = (_Float16*)alloc(1024ull * 256 * 2);
  _Float16* emb16   = (_Float16*)alloc(4096ull * 512 * 2);
  float*    xg      = (float*)   alloc(2ull * 4096 * 1024 * 4);
  _Float16* hbuf    = (_Float16*)alloc(2ull * 64 * 512 * 2);
  float*    cdec    = (float*)   alloc(64ull * 512 * 4);
  unsigned long long* amax = (unsigned long long*)alloc(63ull * 64 * 8);

  // 1) weight conversion f32 -> f16 (once per call; deterministic)
  k_f32_to_f16<<<4096, 256, 0, stream>>>(clf_w,     clfw16,  32000 * 512);
  k_f32_to_f16<<<1024, 256, 0, stream>>>(dec_wih,   dwih16,  2048 * 512);
  k_f32_to_f16<<<1024, 256, 0, stream>>>(dec_whh,   dwhh16,  2048 * 512);
  k_f32_to_f16<<<512,  256, 0, stream>>>(enc_wih_f, ewihf16, 1024 * 512);
  k_f32_to_f16<<<256,  256, 0, stream>>>(enc_whh_f, ewhhf16, 1024 * 256);
  k_f32_to_f16<<<512,  256, 0, stream>>>(enc_wih_b, ewihb16, 1024 * 512);
  k_f32_to_f16<<<256,  256, 0, stream>>>(enc_whh_b, ewhhb16, 1024 * 256);
  k_zero_u64<<<16, 256, 0, stream>>>(amax, 63 * 64);

  // 2) encoder: embedding gather, batched input projection, recurrence
  k_gather_emb<<<4096, 128, 0, stream>>>(x, enc_emb, emb16);
  k_xg<<<dim3(64, 8, 2), 256, 0, stream>>>(emb16, ewihf16, ewihb16, xg);
  k_enc_rnn<<<2, 1024, 0, stream>>>(xg, ewhhf16, ewhhb16, enc_b_f, enc_b_b, hbuf, cdec);

  // 3) decoder: 63 greedy steps (LSTM cell, then classifier + argmax)
  for (int t = 0; t < 63; t++) {
    k_dec_step<<<16, 256, 0, stream>>>(t, dec_emb, dwih16, dwhh16, dec_b, amax, hbuf, cdec);
    k_clf<<<250, 256, 0, stream>>>(t, hbuf, clfw16, clf_b, out, amax);
  }
}
